// StokenAttention_81930796138899
// MI455X (gfx1250) — matrix operand order repaired and checked by reading, not compiled, not run on hardware
//
#include <hip/hip_runtime.h>

typedef __attribute__((ext_vector_type(16))) _Float16 v16h;
typedef __attribute__((ext_vector_type(8)))  _Float16 v8h;
typedef __attribute__((ext_vector_type(8)))  float    v8f;

#define BATCH 8
#define CH    128
#define HSZ   32
#define WSZ   32
#define NPAT  1024
#define PP    64

__device__ __forceinline__ v8f wmma32(v16h a, v16h b, v8f c) {
  return __builtin_amdgcn_wmma_f32_16x16x32_f16(false, a, false, b, (short)0, c, false, false);
}

// A-operand: lane = M row; elements = k in [hv*8, hv*8+7] and [16+hv*8, 16+hv*8+7].
// Caller passes p = row_base + kchunk*32 + hv*8.
__device__ __forceinline__ v16h ld_a(const _Float16* p) {
  v8h lo = *(const v8h*)(p);
  v8h hi = *(const v8h*)(p + 16);
  v16h r;
#pragma unroll
  for (int i = 0; i < 8; ++i) { r[i] = lo[i]; r[8 + i] = hi[i]; }
  return r;
}

// Same but upper K-half (k>=16 of the chunk) is logical zero (K padded to 32).
__device__ __forceinline__ v16h ld_a_lo(const _Float16* p) {
  v8h lo = *(const v8h*)(p);
  v16h r;
#pragma unroll
  for (int i = 0; i < 8; ++i) { r[i] = lo[i]; r[8 + i] = (_Float16)0.f; }
  return r;
}

// B-operand: lane = N; elements = k in [hv*8, hv*8+7] and [16+hv*8, 16+hv*8+7]. Caller passes
// p = btrow_base + kchunk*32 + hv*8 (btrow = row n of B^T, K-contiguous).
__device__ __forceinline__ v16h ld_b(const _Float16* p) { return ld_a(p); }

__device__ __forceinline__ v16h zero16h() {
  v16h r;
#pragma unroll
  for (int i = 0; i < 16; ++i) r[i] = (_Float16)0.f;
  return r;
}

// ---------------------------------------------------------------------------
// K0: f16 weight conversion + zero the padded 34x34 grids (borders matter).
// ---------------------------------------------------------------------------
__global__ void k_init(const float* __restrict__ qkvw_f, const float* __restrict__ projw_f,
                       _Float16* __restrict__ qkvw, _Float16* __restrict__ projw,
                       _Float16* __restrict__ stp, _Float16* __restrict__ sfpad) {
  const int i = blockIdx.x * blockDim.x + threadIdx.x;
  if (i < 384 * CH) qkvw[i] = (_Float16)qkvw_f[i];
  if (i < CH * CH) projw[i] = (_Float16)projw_f[i];
  const int pads = BATCH * 34 * 34 * CH;
  if (i < pads) { stp[i] = (_Float16)0.f; sfpad[i] = (_Float16)0.f; }
}

// ---------------------------------------------------------------------------
// K1: channel LayerNorm; emit pix (P x C, C-contig) and pixT (C x P, P-contig) f16.
// One block per (b, patch). LDS padded to 65 to kill bank conflicts.
// ---------------------------------------------------------------------------
__global__ void k_ln_pix(const float* __restrict__ x, const float* __restrict__ lnw,
                         const float* __restrict__ lnb,
                         _Float16* __restrict__ pix, _Float16* __restrict__ pixT) {
  const int bn = blockIdx.x;
  const int b = bn >> 10, n = bn & (NPAT - 1);
  const int hs = n >> 5, ws = n & 31;
  __shared__ float xs[CH * 65];
  __shared__ float mu[PP];
  __shared__ float rs[PP];
  const int t = threadIdx.x;
  const int xbase = (b * CH) << 16;
  for (int idx = t; idx < CH * PP; idx += 256) {
    const int c = idx >> 6, p = idx & 63;
    const int h = (hs << 3) + (p >> 3), w = (ws << 3) + (p & 7);
    xs[c * 65 + p] = x[xbase + (c << 16) + (h << 8) + w];
  }
  __syncthreads();
  if (t < PP) {
    float s = 0.f;
    for (int c = 0; c < CH; ++c) s += xs[c * 65 + t];
    const float m = s * (1.f / CH);
    float v = 0.f;
    for (int c = 0; c < CH; ++c) { const float d = xs[c * 65 + t] - m; v += d * d; }
    mu[t] = m;
    rs[t] = rsqrtf(v * (1.f / CH) + 1e-6f);
  }
  __syncthreads();
  const int pbase = bn * (PP * CH);
  for (int idx = t; idx < CH * PP; idx += 256) {        // pixT[c][p]
    const int c = idx >> 6, p = idx & 63;
    const float y = (xs[c * 65 + p] - mu[p]) * rs[p] * lnw[c] + lnb[c];
    pixT[pbase + idx] = (_Float16)y;
  }
  for (int idx = t; idx < CH * PP; idx += 256) {        // pix[p][c]
    const int p = idx >> 7, c = idx & 127;
    const float y = (xs[c * 65 + p] - mu[p]) * rs[p] * lnw[c] + lnb[c];
    pix[pbase + idx] = (_Float16)y;
  }
}

// ---------------------------------------------------------------------------
// K2: stoken = patch mean -> padded grid stp (b, 34, 34, C), interior only.
// ---------------------------------------------------------------------------
__global__ void k_stoken(const _Float16* __restrict__ pixT, _Float16* __restrict__ stp) {
  const int bn = blockIdx.x;
  const int b = bn >> 10, n = bn & (NPAT - 1);
  const int hs = n >> 5, ws = n & 31;
  const int c = threadIdx.x;
  const _Float16* r = pixT + bn * (PP * CH) + c * PP;
  float s = 0.f;
#pragma unroll 8
  for (int p = 0; p < PP; ++p) s += (float)r[p];
  stp[((b * 34 + hs + 1) * 34 + (ws + 1)) * CH + c] = (_Float16)(s * (1.f / PP));
}

// ---------------------------------------------------------------------------
// K3: per patch: aff = softmax_9(pix(64x128) x st^T * C^-0.5), colsum, and
// sfp(C x 9) = pixT(128x64) x affT^T — all via WMMA. Block = 4 waves.
// ---------------------------------------------------------------------------
__global__ void k_aff(const _Float16* __restrict__ pix, const _Float16* __restrict__ pixT,
                      const _Float16* __restrict__ stp, _Float16* __restrict__ aff,
                      float* __restrict__ sfp, float* __restrict__ colsum) {
  const int bn = blockIdx.x;
  const int b = bn >> 10, n = bn & (NPAT - 1);
  const int hs = n >> 5, ws = n & 31;
  __shared__ __align__(32) _Float16 stT[16 * CH];   // rows = neighbor k (9 used)
  __shared__ __align__(32) float    S[PP * 17];     // scores/aff, row-padded
  __shared__ __align__(32) _Float16 affT[16 * PP];  // rows = k, cols = p
  const int t = threadIdx.x;
  for (int idx = t; idx < 16 * CH; idx += 128) {
    const int k = idx >> 7, c = idx & 127;
    _Float16 v = (_Float16)0.f;
    if (k < 9) {
      const int dh = k / 3 - 1, dw = k % 3 - 1;
      v = stp[((b * 34 + hs + 1 + dh) * 34 + (ws + 1 + dw)) * CH + c];
    }
    stT[idx] = v;
  }
  __syncthreads();
  const int wave = t >> 5, lane = t & 31;
  const int l16 = lane & 15, hv = lane >> 4;
  const _Float16* pixb = pix + bn * (PP * CH);
  {  // GEMM1: D(64x16) = pix x st^T, K = 128 (4 chunks)
    v8f acc = {};
    const _Float16* arow = pixb + (wave * 16 + l16) * CH + hv * 8;
    const _Float16* brow = stT + l16 * CH + hv * 8;
#pragma unroll
    for (int kc = 0; kc < 4; ++kc)
      acc = wmma32(ld_a(arow + kc * 32), ld_b(brow + kc * 32), acc);
    const float scale = 0.08838834764831845f;  // 128^-0.5
#pragma unroll
    for (int rr = 0; rr < 8; ++rr) {
      const int m = wave * 16 + hv * 8 + rr;
      S[m * 17 + l16] = acc[rr] * scale;
    }
  }
  __syncthreads();
  if (t < PP) {  // softmax over the 9 neighbors
    float mx = -1e30f;
#pragma unroll
    for (int k = 0; k < 9; ++k) mx = fmaxf(mx, S[t * 17 + k]);
    float e[9];
    float sum = 0.f;
#pragma unroll
    for (int k = 0; k < 9; ++k) { e[k] = __expf(S[t * 17 + k] - mx); sum += e[k]; }
    const float inv = 1.f / sum;
#pragma unroll
    for (int k = 0; k < 9; ++k) {
      const float a = e[k] * inv;
      S[t * 17 + k] = a;
      affT[k * PP + t] = (_Float16)a;
    }
#pragma unroll
    for (int k = 9; k < 16; ++k) { S[t * 17 + k] = 0.f; affT[k * PP + t] = (_Float16)0.f; }
  }
  __syncthreads();
  for (int idx = t; idx < PP * 16; idx += 128) {  // aff global (p-major, k-contig)
    const int p = idx >> 4, k = idx & 15;
    aff[bn * (PP * 16) + idx] = (_Float16)S[p * 17 + k];
  }
  if (t < 16) {
    float s = 0.f;
    for (int p = 0; p < PP; ++p) s += S[p * 17 + t];
    colsum[bn * 16 + t] = s;
  }
  // GEMM2: sfp[c][k] = sum_p pixT[c][p]*affT[k][p]; M=128, N=16, K=64 (2 chunks)
  const _Float16* pixTb = pixT + bn * (PP * CH);
#pragma unroll
  for (int mi = 0; mi < 2; ++mi) {
    const int mt = wave + mi * 4;
    v8f acc = {};
    const _Float16* arow = pixTb + (mt * 16 + l16) * PP + hv * 8;
    const _Float16* brow = affT + l16 * PP + hv * 8;
#pragma unroll
    for (int kc = 0; kc < 2; ++kc)
      acc = wmma32(ld_a(arow + kc * 32), ld_b(brow + kc * 32), acc);
#pragma unroll
    for (int rr = 0; rr < 8; ++rr) {
      const int c = mt * 16 + hv * 8 + rr;
      sfp[(bn * CH + c) * 16 + l16] = acc[rr];
    }
  }
}

// ---------------------------------------------------------------------------
// K4: fold3 (as a gather) + divide by folded aff_sum -> tok f16 (B,1024,C).
// ---------------------------------------------------------------------------
__global__ void k_fold(const float* __restrict__ sfp, const float* __restrict__ colsum,
                       _Float16* __restrict__ tok) {
  const int bt = blockIdx.x;
  const int b = bt >> 10, n = bt & (NPAT - 1);
  const int hs = n >> 5, ws = n & 31;
  const int c = threadIdx.x;
  float asum = 0.f, val = 0.f;
#pragma unroll
  for (int k = 0; k < 9; ++k) {
    const int dh = k / 3 - 1, dw = k % 3 - 1;
    const int sh = hs - dh, sw = ws - dw;
    if (sh < 0 || sh >= HSZ || sw < 0 || sw >= WSZ) continue;
    const int sn = (b << 10) + (sh << 5) + sw;
    asum += colsum[sn * 16 + k];
    val += sfp[(sn * CH + c) * 16 + k];
  }
  tok[bt * CH + c] = (_Float16)(val / (asum + 1e-12f));
}

// ---------------------------------------------------------------------------
// K5: qkv = tok @ qkv_w^T (M=1024/batch, N=384, K=128). q,k written hd-padded
// to 32 (zeros), v written transposed (hd x N) for the PV WMMA B-operand.
// ---------------------------------------------------------------------------
__global__ void k_qkv(const _Float16* __restrict__ tok, const _Float16* __restrict__ qkvw,
                      _Float16* __restrict__ q, _Float16* __restrict__ kmat,
                      _Float16* __restrict__ vT) {
  const int blk = blockIdx.x;
  const int b = blk >> 4, mb = blk & 15;
  const int t = threadIdx.x;
  const int wave = t >> 5, lane = t & 31, l16 = lane & 15, hv = lane >> 4;
  const int row0 = mb * 64 + wave * 16;
  const _Float16* arow = tok + (b * 1024 + row0 + l16) * CH + hv * 8;
  v16h a[4];
#pragma unroll
  for (int kc = 0; kc < 4; ++kc) a[kc] = ld_a(arow + kc * 32);
  for (int nt = 0; nt < 24; ++nt) {
    v8f acc = {};
    const _Float16* brow = qkvw + (nt * 16 + l16) * CH + hv * 8;
#pragma unroll
    for (int kc = 0; kc < 4; ++kc) acc = wmma32(a[kc], ld_b(brow + kc * 32), acc);
    const int mat = nt >> 3;   // uniform per tile: 0=q 1=k 2=v
    const int head = nt & 7;   // uniform per tile
    const int hd = l16;
#pragma unroll
    for (int r = 0; r < 8; ++r) {
      const int row = row0 + hv * 8 + r;
      const _Float16 val = (_Float16)acc[r];
      if (mat == 0) {
        const int base = ((b * 8 + head) * 1024 + row) * 32;
        q[base + hd] = val;
        q[base + hd + 16] = (_Float16)0.f;
      } else if (mat == 1) {
        const int base = ((b * 8 + head) * 1024 + row) * 32;
        kmat[base + hd] = val;
        kmat[base + hd + 16] = (_Float16)0.f;
      } else {
        vT[((b * 8 + head) * 16 + hd) * 1024 + row] = val;
      }
    }
  }
}

// ---------------------------------------------------------------------------
// K6: flash-style attention. Block = (b, head, 64-row q block), 4 waves.
// Online softmax: row stats reduced over 16-lane halves with wave32 shuffles;
// P tiles go through LDS to become the A-operand of the PV WMMA.
// ---------------------------------------------------------------------------
__global__ void k_attn(const _Float16* __restrict__ q, const _Float16* __restrict__ kmat,
                       const _Float16* __restrict__ vT, _Float16* __restrict__ aout) {
  const int blk = blockIdx.x;
  const int qb = blk & 15, bh = blk >> 4;
  const int b = bh >> 3, head = bh & 7;
  const int t = threadIdx.x;
  const int wave = t >> 5, lane = t & 31, l16 = lane & 15, hv = lane >> 4;
  __shared__ __align__(32) _Float16 P[4][16][32];
  const int row0 = qb * 64 + wave * 16;
  const v16h qa = ld_a(q + (bh * 1024 + row0 + l16) * 32 + hv * 8);
  v8f o = {};
  float mrun[8], lrun[8];
#pragma unroll
  for (int r = 0; r < 8; ++r) { mrun[r] = -1e30f; lrun[r] = 0.f; }
  const _Float16* kbase = kmat + bh * (1024 * 32);
  const _Float16* vbase = vT + bh * (16 * 1024);
  for (int kt = 0; kt < 32; ++kt) {            // 32 chunks of 32 keys
    v8f s0 = {}, s1 = {};
    s0 = wmma32(qa, ld_b(kbase + (kt * 32 + l16) * 32 + hv * 8), s0);
    s1 = wmma32(qa, ld_b(kbase + (kt * 32 + 16 + l16) * 32 + hv * 8), s1);
#pragma unroll
    for (int r = 0; r < 8; ++r) {
      const float a0 = s0[r] * 0.25f;          // hd^-0.5
      const float a1 = s1[r] * 0.25f;
      float mx = fmaxf(a0, a1);
      mx = fmaxf(mx, __shfl_xor(mx, 1, 32));
      mx = fmaxf(mx, __shfl_xor(mx, 2, 32));
      mx = fmaxf(mx, __shfl_xor(mx, 4, 32));
      mx = fmaxf(mx, __shfl_xor(mx, 8, 32));
      const float mnew = fmaxf(mrun[r], mx);
      const float corr = __expf(mrun[r] - mnew);
      const float p0 = __expf(a0 - mnew);
      const float p1 = __expf(a1 - mnew);
      float ts = p0 + p1;
      ts += __shfl_xor(ts, 1, 32);
      ts += __shfl_xor(ts, 2, 32);
      ts += __shfl_xor(ts, 4, 32);
      ts += __shfl_xor(ts, 8, 32);
      lrun[r] = lrun[r] * corr + ts;
      mrun[r] = mnew;
      o[r] = o[r] * corr;
      const int m = hv * 8 + r;
      P[wave][m][l16] = (_Float16)p0;
      P[wave][m][l16 + 16] = (_Float16)p1;
    }
    __syncthreads();
    o = wmma32(ld_a(&P[wave][l16][hv * 8]),
               ld_b(vbase + l16 * 1024 + kt * 32 + hv * 8), o);
    __syncthreads();
  }
#pragma unroll
  for (int r = 0; r < 8; ++r) {
    const int row = row0 + hv * 8 + r;
    aout[(b * 1024 + row) * CH + head * 16 + l16] = (_Float16)(o[r] / lrun[r]);
  }
}

// ---------------------------------------------------------------------------
// K7: proj GEMM + bias -> padded sf grid (interior of sfpad).
// ---------------------------------------------------------------------------
__global__ void k_proj(const _Float16* __restrict__ aout, const _Float16* __restrict__ projw,
                       const float* __restrict__ projb, _Float16* __restrict__ sfpad) {
  const int blk = blockIdx.x;
  const int b = blk >> 4, mb = blk & 15;
  const int t = threadIdx.x;
  const int wave = t >> 5, lane = t & 31, l16 = lane & 15, hv = lane >> 4;
  const int row0 = mb * 64 + wave * 16;
  const _Float16* arow = aout + (b * 1024 + row0 + l16) * CH + hv * 8;
  v16h a[4];
#pragma unroll
  for (int kc = 0; kc < 4; ++kc) a[kc] = ld_a(arow + kc * 32);
  for (int nt = 0; nt < 8; ++nt) {
    v8f acc = {};
    const _Float16* brow = projw + (nt * 16 + l16) * CH + hv * 8;
#pragma unroll
    for (int kc = 0; kc < 4; ++kc) acc = wmma32(a[kc], ld_b(brow + kc * 32), acc);
    const int j = nt * 16 + l16;
    const float bias = projb[j];
#pragma unroll
    for (int r = 0; r < 8; ++r) {
      const int row = row0 + hv * 8 + r;
      const int hh = row >> 5, ww = row & 31;
      sfpad[((b * 34 + hh + 1) * 34 + (ww + 1)) * CH + j] = (_Float16)(acc[r] + bias);
    }
  }
}

// ---------------------------------------------------------------------------
// K8: out[c][p] = sum_k st2[c][k]*aff[p][k]; M=128, N=64, K=16 (padded to 32).
// st2 built in LDS c-major from 9 contiguous channel vectors of sfpad.
// ---------------------------------------------------------------------------
__global__ void k_out(const _Float16* __restrict__ sfpad, const _Float16* __restrict__ aff,
                      float* __restrict__ out) {
  const int bn = blockIdx.x;
  const int b = bn >> 10, n = bn & (NPAT - 1);
  const int hs = n >> 5, ws = n & 31;
  __shared__ __align__(32) _Float16 st2[CH * 16];
  const int t = threadIdx.x;
  for (int idx = t; idx < CH * 16; idx += 128) st2[idx] = (_Float16)0.f;
  __syncthreads();
  for (int idx = t; idx < 9 * CH; idx += 128) {
    const int k = idx >> 7, c = idx & 127;
    const int dh = k / 3 - 1, dw = k % 3 - 1;
    st2[c * 16 + k] = sfpad[((b * 34 + hs + 1 + dh) * 34 + (ws + 1 + dw)) * CH + c];
  }
  __syncthreads();
  const int wave = t >> 5, lane = t & 31, l16 = lane & 15, hv = lane >> 4;
  const _Float16* affb = aff + bn * (PP * 16);
  const v16h bz = zero16h();
#pragma unroll
  for (int mi = 0; mi < 2; ++mi) {
    const int mt = wave * 2 + mi;
    const v16h a = ld_a_lo(&st2[(mt * 16 + l16) * 16 + hv * 8]);
#pragma unroll
    for (int nt = 0; nt < 4; ++nt) {
      v8f acc = {};
      const v16h bfr = ld_a_lo(affb + (nt * 16 + l16) * 16 + hv * 8);  // k>=16 is zero
      acc = wmma32(a, bfr, acc);
#pragma unroll
      for (int r = 0; r < 8; ++r) {
        const int c = mt * 16 + hv * 8 + r;
        const int p = nt * 16 + l16;
        const int h = (hs << 3) + (p >> 3), w = (ws << 3) + (p & 7);
        out[((b * CH + c) << 16) + (h << 8) + w] = acc[r];
      }
    }
  }
}

// ---------------------------------------------------------------------------
extern "C" void kernel_launch(void* const* d_in, const int* in_sizes, int n_in,
                              void* d_out, int out_size, void* d_ws, size_t ws_size,
                              hipStream_t stream) {
  (void)in_sizes; (void)n_in; (void)out_size; (void)ws_size;
  const float* x       = (const float*)d_in[0];
  const float* lnw     = (const float*)d_in[1];
  const float* lnb     = (const float*)d_in[2];
  const float* qkvw_f  = (const float*)d_in[3];
  const float* projw_f = (const float*)d_in[4];
  const float* projb   = (const float*)d_in[5];
  float* out = (float*)d_out;

  char* ws = (char*)d_ws;
  size_t off = 0;
  auto take = [&](size_t bytes) -> char* {
    char* p = ws + off;
    off += (bytes + 255) & ~(size_t)255;
    return p;
  };
  _Float16* pix     = (_Float16*)take((size_t)BATCH * NPAT * PP * CH * 2);
  _Float16* pixT    = (_Float16*)take((size_t)BATCH * NPAT * PP * CH * 2);
  _Float16* stp     = (_Float16*)take((size_t)BATCH * 34 * 34 * CH * 2);
  _Float16* aff     = (_Float16*)take((size_t)BATCH * NPAT * PP * 16 * 2);
  float*    sfp     = (float*)take((size_t)BATCH * NPAT * CH * 16 * 4);
  float*    colsum  = (float*)take((size_t)BATCH * NPAT * 16 * 4);
  _Float16* tok     = (_Float16*)take((size_t)BATCH * NPAT * CH * 2);
  _Float16* qkvw16  = (_Float16*)take((size_t)3 * CH * CH * 2);
  _Float16* projw16 = (_Float16*)take((size_t)CH * CH * 2);
  _Float16* qbuf    = (_Float16*)take((size_t)BATCH * 8 * 1024 * 32 * 2);
  _Float16* kbuf    = (_Float16*)take((size_t)BATCH * 8 * 1024 * 32 * 2);
  _Float16* vtbuf   = (_Float16*)take((size_t)BATCH * 8 * 16 * 1024 * 2);
  _Float16* aoutb   = (_Float16*)take((size_t)BATCH * NPAT * CH * 2);
  _Float16* sfpad   = (_Float16*)take((size_t)BATCH * 34 * 34 * CH * 2);

  k_init  <<<4624, 256, 0, stream>>>(qkvw_f, projw_f, qkvw16, projw16, stp, sfpad);
  k_ln_pix<<<BATCH * NPAT, 256, 0, stream>>>(x, lnw, lnb, pix, pixT);
  k_stoken<<<BATCH * NPAT, 128, 0, stream>>>(pixT, stp);
  k_aff   <<<BATCH * NPAT, 128, 0, stream>>>(pix, pixT, stp, aff, sfp, colsum);
  k_fold  <<<BATCH * NPAT, 128, 0, stream>>>(sfp, colsum, tok);
  k_qkv   <<<BATCH * 16, 128, 0, stream>>>(tok, qkvw16, qbuf, kbuf, vtbuf);
  k_attn  <<<BATCH * 8 * 16, 128, 0, stream>>>(qbuf, kbuf, vtbuf, aoutb);
  k_proj  <<<BATCH * 16, 128, 0, stream>>>(aoutb, projw16, projb, sfpad);
  k_out   <<<BATCH * NPAT, 128, 0, stream>>>(sfpad, aff, out);
}
